// MultiHeadSelfAttention_7533372637525
// MI455X (gfx1250) — compile-verified, matrix operand order checked by reading, not run on hardware
//
#include <hip/hip_runtime.h>
#include <hip/hip_bf16.h>
#include <math.h>

// MI455X / gfx1250: wave32, WMMA 16x16x32 f16 (f32 accum), TDM tile staging.
// MHA forward, ~137 GFLOP, matrix-pipe bound.

typedef __attribute__((ext_vector_type(16))) _Float16 v16h;
typedef __attribute__((ext_vector_type(8)))  float    v8f;
typedef unsigned int u32x4 __attribute__((ext_vector_type(4)));
typedef int          i32x8 __attribute__((ext_vector_type(8)));
typedef int          i32x4 __attribute__((ext_vector_type(4)));

#define NB  4
#define NS  2048
#define ND  1024
#define NH  16
#define NDH 64

union FragU { v16h h; uint4 q[2]; };

// ---------------------------------------------------------------------------
// TDM: async 2D tile (rows x elems, f16) from global -> LDS with HW padding.
// Issued per-wave; caller guards with a wave-uniform branch and waits with
// s_wait_tensorcnt. D# packing per CDNA5 ISA ch.8 (group0 128b, group1 256b).
// This toolchain's builtin takes 6 args (extra int32x8 group, zero for 2D).
// ---------------------------------------------------------------------------
__device__ __forceinline__ void tdm_load_2d_f16(unsigned lds_off, const _Float16* gptr,
                                                unsigned tile_x, unsigned tile_rows,
                                                unsigned row_stride_elems,
                                                unsigned pad_interval_code,
                                                unsigned pad_amount_code) {
  unsigned long long ga = (unsigned long long)(uintptr_t)gptr;
  u32x4 g0;
  g0[0] = 1u;                                   // count=1 valid user descriptor
  g0[1] = lds_off;                              // lds_addr (bytes)
  g0[2] = (unsigned)ga;                         // global_addr[31:0]
  g0[3] = (unsigned)((ga >> 32) & 0x01FFFFFFu)  // global_addr[56:32]
          | (2u << 30);                         // type=2 ("image")
  const unsigned td0 = 0x7FFFFFFFu, td1 = 0x7FFFFFFFu;  // no OOB clipping
  i32x8 g1;
  g1[0] = (int)((1u << 16)                      // data_size = 2 bytes
                | (1u << 20)                    // pad_enable
                | (pad_interval_code << 22) | (pad_amount_code << 25));
  g1[1] = (int)((td0 & 0xFFFFu) << 16);         // tensor_dim0 lo16 (atomic addr = 0)
  g1[2] = (int)((td0 >> 16) | ((td1 & 0xFFFFu) << 16));
  g1[3] = (int)((td1 >> 16) | (tile_x << 16));  // tile_dim0
  g1[4] = (int)(tile_rows & 0xFFFFu);           // tile_dim1 (tile_dim2 = 0)
  g1[5] = (int)row_stride_elems;                // tensor_dim0_stride lo32
  g1[6] = 0;
  g1[7] = 0;
  i32x4 z4 = {0, 0, 0, 0};                      // groups 2/3 unused (2D)
  i32x8 z8 = {0, 0, 0, 0, 0, 0, 0, 0};          // extra group (clang-23 form)
  __builtin_amdgcn_tensor_load_to_lds(g0, g1, z4, z4, z8, 0);
}

// ---------------------------------------------------------------------------
__global__ __launch_bounds__(256) void cvt_f32_f16(const float* __restrict__ in,
                                                   _Float16* __restrict__ out, int n) {
  int i = blockIdx.x * 256 + threadIdx.x;
  if (i < n) out[i] = (_Float16)in[i];
}

// ---------------------------------------------------------------------------
// C[m,n] = sum_k A[m,k] * W[n,k]. Block tile 128x128, 8 waves (32x64 each),
// K-step 32. Tiles staged by TDM with HW-padded 40-half pitch (80B rows:
// 16-lane b128 frag loads hit 16 distinct bank quads -> conflict-free).
// ---------------------------------------------------------------------------
template <bool STOREHALF>
__global__ __launch_bounds__(256) void gemm_xwT(const _Float16* __restrict__ A,
                                                const _Float16* __restrict__ W,
                                                float* __restrict__ Cf,
                                                _Float16* __restrict__ Ch,
                                                int M, int N, int K) {
  __shared__ _Float16 lA[128 * 40];
  __shared__ _Float16 lW[128 * 40];

  const int tid  = threadIdx.x;
  const int lane = tid & 31;
  const int wid  = tid >> 5;
  const int wm   = wid & 3;
  const int wn   = wid >> 2;
  const int mBase = blockIdx.x * 128;
  const int nBase = blockIdx.y * 128;
  const unsigned ldsA = (unsigned)(uintptr_t)(&lA[0]);
  const unsigned ldsW = (unsigned)(uintptr_t)(&lW[0]);

  v8f acc[2][4];
#pragma unroll
  for (int i = 0; i < 2; ++i)
#pragma unroll
    for (int j = 0; j < 4; ++j)
#pragma unroll
      for (int r = 0; r < 8; ++r) acc[i][j][r] = 0.0f;

  for (int k0 = 0; k0 < K; k0 += 32) {
    __syncthreads();  // previous tile fully consumed
    if (wid == 0) {
      // 128 rows x 32 halves, pad 4 DWORDs every 16 DWORDs -> 40-half pitch
      tdm_load_2d_f16(ldsA, A + (size_t)mBase * K + k0, 32, 128, K, 3, 3);
      tdm_load_2d_f16(ldsW, W + (size_t)nBase * K + k0, 32, 128, K, 3, 3);
      __builtin_amdgcn_s_wait_tensorcnt(0);
    }
    __syncthreads();  // tiles visible to all waves

    const int rl = lane & 15;
    const int c0 = (lane & 16) ? 8 : 0;

    v16h af[2], bf[4];
#pragma unroll
    for (int i = 0; i < 2; ++i) {
      int r = wm * 32 + i * 16 + rl;
      FragU u;
      u.q[0] = *(const uint4*)(&lA[r * 40 + c0]);
      u.q[1] = *(const uint4*)(&lA[r * 40 + c0 + 16]);
      af[i] = u.h;
    }
#pragma unroll
    for (int j = 0; j < 4; ++j) {
      int r = wn * 64 + j * 16 + rl;
      FragU u;
      u.q[0] = *(const uint4*)(&lW[r * 40 + c0]);
      u.q[1] = *(const uint4*)(&lW[r * 40 + c0 + 16]);
      bf[j] = u.h;
    }
#pragma unroll
    for (int i = 0; i < 2; ++i)
#pragma unroll
      for (int j = 0; j < 4; ++j)
        acc[i][j] = __builtin_amdgcn_wmma_f32_16x16x32_f16(
            false, af[i], false, bf[j], (short)0, acc[i][j], false, false);
  }

  const int rl = lane & 15;
  const int hb = (lane & 16) ? 8 : 0;
#pragma unroll
  for (int i = 0; i < 2; ++i)
#pragma unroll
    for (int j = 0; j < 4; ++j)
#pragma unroll
      for (int r = 0; r < 8; ++r) {
        int m = mBase + wm * 32 + i * 16 + r + hb;
        int n = nBase + wn * 64 + j * 16 + rl;
        if (STOREHALF)
          Ch[(size_t)m * N + n] = (_Float16)acc[i][j][r];
        else
          Cf[(size_t)m * N + n] = acc[i][j][r];
      }
}

// ---------------------------------------------------------------------------
// RoPE (interleaved pairs) on Q,K + [B,S,H*Dh] -> [B,H,S,Dh]; V permute-only.
// ---------------------------------------------------------------------------
__global__ __launch_bounds__(256) void rope_permute(const _Float16* __restrict__ qg,
                                                    const _Float16* __restrict__ kg,
                                                    const _Float16* __restrict__ vg,
                                                    const int* __restrict__ pos,
                                                    _Float16* __restrict__ qt,
                                                    _Float16* __restrict__ kt,
                                                    _Float16* __restrict__ vt) {
  const int PPM = NB * NS * ND / 2;
  long long idx = (long long)blockIdx.x * 256 + threadIdx.x;
  int mat = (int)(idx / PPM);
  int pr  = (int)(idx % PPM);
  int d2 = pr % (ND / 2);
  int s  = (pr / (ND / 2)) % NS;
  int b  = pr / ((ND / 2) * NS);
  int e0  = d2 * 2;
  int h   = e0 >> 6;
  int dh0 = e0 & 63;

  const _Float16* src = (mat == 0) ? qg : (mat == 1) ? kg : vg;
  _Float16*       dst = (mat == 0) ? qt : (mat == 1) ? kt : vt;

  size_t si = ((size_t)(b * NS + s)) * ND + e0;
  float a = (float)src[si];
  float c = (float)src[si + 1];
  float o0 = a, o1 = c;
  if (mat < 2) {
    float p   = (float)pos[s];
    float inv = __expf((float)dh0 * (-9.210340371976184f / 64.0f));
    float ang = p * inv;
    float sn, cs;
    sincosf(ang, &sn, &cs);  // full range reduction (ang up to ~2047 rad)
    o0 = a * cs - c * sn;
    o1 = c * cs + a * sn;
  }
  size_t di = ((size_t)((b * NH + h) * NS + s)) * NDH + dh0;
  dst[di]     = (_Float16)o0;
  dst[di + 1] = (_Float16)o1;
}

// ---------------------------------------------------------------------------
// Causal flash attention. 128 threads = 4 waves; block = (b, h, 64-query
// tile); wave = 16 queries. 128-key tiles: K staged by TDM (72-half pitch),
// V transposed by the waves while the TDM load is in flight. Fragment loads
// batched 4-deep ahead of the WMMAs so DS latency hides under the matrix pipe.
// ---------------------------------------------------------------------------
__global__ __launch_bounds__(128) void flash_attn(const _Float16* __restrict__ Qt,
                                                  const _Float16* __restrict__ Kt,
                                                  const _Float16* __restrict__ Vt,
                                                  _Float16* __restrict__ AT) {
  __shared__ _Float16 lK[128 * 72];      // 18 KB : K tile [key][dh], TDM-padded
  __shared__ _Float16 lVT[NDH * 136];    // 17 KB : V tile transposed [dh][key]
  __shared__ _Float16 lP[4 * 16 * 136];  // 17 KB : per-wave P scratch [q][key]

  const int tid  = threadIdx.x;
  const int lane = tid & 31;
  const int wid  = tid >> 5;
  const int qtile = blockIdx.x;
  const int h     = blockIdx.y;
  const int b     = blockIdx.z;
  const int qbase = qtile * 64 + wid * 16;
  const size_t headOff = ((size_t)(b * NH + h)) * NS * NDH;
  const unsigned ldsK = (unsigned)(uintptr_t)(&lK[0]);

  const int rl = lane & 15;
  const int c0 = (lane & 16) ? 8 : 0;
  const int hb = (lane & 16) ? 8 : 0;

  v16h qf[2];
  {
    const _Float16* qp = Qt + headOff + (size_t)(qbase + rl) * NDH;
#pragma unroll
    for (int ks = 0; ks < 2; ++ks) {
      FragU u;
      u.q[0] = *(const uint4*)(qp + ks * 32 + c0);
      u.q[1] = *(const uint4*)(qp + ks * 32 + c0 + 16);
      qf[ks] = u.h;
    }
  }

  float mi[8], li[8];
#pragma unroll
  for (int r = 0; r < 8; ++r) { mi[r] = -1e30f; li[r] = 0.0f; }
  v8f o[4];
#pragma unroll
  for (int j = 0; j < 4; ++j)
#pragma unroll
    for (int r = 0; r < 8; ++r) o[j][r] = 0.0f;

  _Float16* pw = &lP[wid * 16 * 136];
  const int ntiles = (qtile >> 1) + 1;

  for (int kt = 0; kt < ntiles; ++kt) {
    const int ktbase = kt * 128;
    __syncthreads();  // previous tiles fully consumed
    if (wid == 0) {
      // K: 128 rows x 64 halves, pad 4 DWORDs every 32 DWORDs -> 72-half pitch
      tdm_load_2d_f16(ldsK, Kt + headOff + (size_t)ktbase * NDH, 64, 128, NDH, 4, 3);
    }
    // V transpose copy overlaps with in-flight TDM K load
#pragma unroll
    for (int p = 0; p < 8; ++p) {
      int c   = tid + p * 128;
      int row = c >> 3;
      int col = (c & 7) * 8;
      union { uint4 u; _Float16 hh[8]; } t;
      t.u = *(const uint4*)(Vt + headOff + (size_t)(ktbase + row) * NDH + col);
#pragma unroll
      for (int j = 0; j < 8; ++j) lVT[(col + j) * 136 + row] = t.hh[j];
    }
    if (wid == 0) __builtin_amdgcn_s_wait_tensorcnt(0);
    __syncthreads();

    // S = Q K^T : 16x128 scores; batch 4 B-frag loads per WMMA group
    v8f sf[8];
#pragma unroll
    for (int nf = 0; nf < 8; ++nf)
#pragma unroll
      for (int r = 0; r < 8; ++r) sf[nf][r] = 0.0f;
#pragma unroll
    for (int ks = 0; ks < 2; ++ks) {
      const int cc = ks * 32 + c0;
#pragma unroll
      for (int g = 0; g < 2; ++g) {
        v16h kf[4];
#pragma unroll
        for (int t = 0; t < 4; ++t) {
          int r = (g * 4 + t) * 16 + rl;
          FragU u;
          u.q[0] = *(const uint4*)(&lK[r * 72 + cc]);
          u.q[1] = *(const uint4*)(&lK[r * 72 + cc + 16]);
          kf[t] = u.h;
        }
#pragma unroll
        for (int t = 0; t < 4; ++t)
          sf[g * 4 + t] = __builtin_amdgcn_wmma_f32_16x16x32_f16(
              false, qf[ks], false, kf[t], (short)0, sf[g * 4 + t], false, false);
      }
    }

    // scale + causal mask (element (r,lane): m = r+hb, n = lane&15)
    const int qrow_lo = qbase + hb;
    const int kcol    = ktbase + rl;
#pragma unroll
    for (int nf = 0; nf < 8; ++nf)
#pragma unroll
      for (int r = 0; r < 8; ++r) {
        float v = sf[nf][r] * 0.125f;  // 1/sqrt(64)
        sf[nf][r] = (kcol + nf * 16 <= qrow_lo + r) ? v : -1e30f;
      }

    // online softmax per row (row elements live across a 16-lane group)
#pragma unroll
    for (int r = 0; r < 8; ++r) {
      float mx = mi[r];
#pragma unroll
      for (int nf = 0; nf < 8; ++nf) mx = fmaxf(mx, sf[nf][r]);
#pragma unroll
      for (int off = 8; off > 0; off >>= 1) mx = fmaxf(mx, __shfl_xor(mx, off, 32));
      float alpha = __expf(mi[r] - mx);
      float rs = 0.0f;
#pragma unroll
      for (int nf = 0; nf < 8; ++nf) {
        float pv = __expf(sf[nf][r] - mx);
        sf[nf][r] = pv;
        rs += pv;
      }
#pragma unroll
      for (int off = 8; off > 0; off >>= 1) rs += __shfl_xor(rs, off, 32);
      li[r] = li[r] * alpha + rs;
      mi[r] = mx;
#pragma unroll
      for (int j = 0; j < 4; ++j) o[j][r] *= alpha;
    }

    // P (C layout) -> per-wave LDS scratch as f16 (A-fragment source layout)
#pragma unroll
    for (int nf = 0; nf < 8; ++nf)
#pragma unroll
      for (int r = 0; r < 8; ++r)
        pw[(r + hb) * 136 + nf * 16 + rl] = (_Float16)sf[nf][r];
    __syncthreads();

    // O += P V : batch A frag + 4 B frags before the 4 WMMAs
#pragma unroll
    for (int ks = 0; ks < 4; ++ks) {
      int cc = ks * 32 + c0;
      FragU ua;
      ua.q[0] = *(const uint4*)(&pw[rl * 136 + cc]);
      ua.q[1] = *(const uint4*)(&pw[rl * 136 + cc + 16]);
      v16h vb[4];
#pragma unroll
      for (int j = 0; j < 4; ++j) {
        int vr = j * 16 + rl;
        FragU ub;
        ub.q[0] = *(const uint4*)(&lVT[vr * 136 + cc]);
        ub.q[1] = *(const uint4*)(&lVT[vr * 136 + cc + 16]);
        vb[j] = ub.h;
      }
#pragma unroll
      for (int j = 0; j < 4; ++j)
        o[j] = __builtin_amdgcn_wmma_f32_16x16x32_f16(false, ua.h, false, vb[j],
                                                      (short)0, o[j], false, false);
    }
  }

  // normalize, write f16 [B,S,D] for the wo GEMM
#pragma unroll
  for (int j = 0; j < 4; ++j)
#pragma unroll
    for (int r = 0; r < 8; ++r) {
      int qq = qbase + r + hb;
      int e  = h * NDH + j * 16 + rl;
      AT[((size_t)(b * NS + qq)) * ND + e] = (_Float16)(o[j][r] / li[r]);
    }
}

// ---------------------------------------------------------------------------
extern "C" void kernel_launch(void* const* d_in, const int* in_sizes, int n_in,
                              void* d_out, int out_size, void* d_ws, size_t ws_size,
                              hipStream_t stream) {
  (void)in_sizes; (void)n_in; (void)out_size; (void)ws_size;
  const float* x   = (const float*)d_in[0];
  const int*   pos = (const int*)d_in[1];
  const float* wq  = (const float*)d_in[2];
  const float* wk  = (const float*)d_in[3];
  const float* wv  = (const float*)d_in[4];
  const float* wo  = (const float*)d_in[5];
  float* out = (float*)d_out;
  char*  ws  = (char*)d_ws;

  const size_t BSD = (size_t)NB * NS * ND;
  const size_t DD  = (size_t)ND * ND;

  size_t off = 0;
  auto nxt = [&](size_t bytes) {
    size_t cur = off;
    off += (bytes + 255) & ~(size_t)255;
    return cur;
  };
  _Float16* xh  = (_Float16*)(ws + nxt(BSD * 2));
  _Float16* wqh = (_Float16*)(ws + nxt(DD * 2));
  _Float16* wkh = (_Float16*)(ws + nxt(DD * 2));
  _Float16* wvh = (_Float16*)(ws + nxt(DD * 2));
  _Float16* woh = (_Float16*)(ws + nxt(DD * 2));
  _Float16* qg  = (_Float16*)(ws + nxt(BSD * 2));
  _Float16* kg  = (_Float16*)(ws + nxt(BSD * 2));
  _Float16* vg  = (_Float16*)(ws + nxt(BSD * 2));
  _Float16* qt  = (_Float16*)(ws + nxt(BSD * 2));
  _Float16* ktb = (_Float16*)(ws + nxt(BSD * 2));
  _Float16* vtb = (_Float16*)(ws + nxt(BSD * 2));
  _Float16* at  = (_Float16*)(ws + nxt(BSD * 2));

  cvt_f32_f16<<<(unsigned)((BSD + 255) / 256), 256, 0, stream>>>(x, xh, (int)BSD);
  cvt_f32_f16<<<(unsigned)((DD + 255) / 256), 256, 0, stream>>>(wq, wqh, (int)DD);
  cvt_f32_f16<<<(unsigned)((DD + 255) / 256), 256, 0, stream>>>(wk, wkh, (int)DD);
  cvt_f32_f16<<<(unsigned)((DD + 255) / 256), 256, 0, stream>>>(wv, wvh, (int)DD);
  cvt_f32_f16<<<(unsigned)((DD + 255) / 256), 256, 0, stream>>>(wo, woh, (int)DD);

  const int M = NB * NS;
  dim3 gg(M / 128, ND / 128);
  gemm_xwT<true><<<gg, 256, 0, stream>>>(xh, wqh, nullptr, qg, M, ND, ND);
  gemm_xwT<true><<<gg, 256, 0, stream>>>(xh, wkh, nullptr, kg, M, ND, ND);
  gemm_xwT<true><<<gg, 256, 0, stream>>>(xh, wvh, nullptr, vg, M, ND, ND);

  const size_t npairs = 3 * BSD / 2;
  rope_permute<<<(unsigned)(npairs / 256), 256, 0, stream>>>(qg, kg, vg, pos,
                                                             qt, ktb, vtb);

  flash_attn<<<dim3(NS / 64, NH, NB), 128, 0, stream>>>(qt, ktb, vtb, at);

  gemm_xwT<false><<<gg, 256, 0, stream>>>(at, woh, out, nullptr, M, ND, ND);
}
